// HyperMixing_75866302316728
// MI455X (gfx1250) — matrix operand order repaired and checked by reading, not compiled, not run on hardware
//
#include <hip/hip_runtime.h>

// ---------------------------------------------------------------------------
// HyperMixing for MI455X (gfx1250).
// bf16 WMMA (v_wmma_f32_16x16x32_bf16), fp32 accumulation.
// W1/W2 intermediates in bf16 workspace (~134 MB, mostly L2-resident).
// k_mix1 K-split 4x for WGP utilization; k_mix2 stages tiles with
// global_load_async_to_lds_b128 (ASYNCcnt) instead of VGPR round-trips.
// ---------------------------------------------------------------------------

typedef __bf16 bf16;
typedef __attribute__((ext_vector_type(16))) __bf16 v16bf;
typedef __attribute__((ext_vector_type(8)))  float  v8f;

constexpr int kD = 512, kH = 8, kDH = 64, kHYPH = 256, kB = 8, kS = 2048;
constexpr int kSeg = 4, kSegLen = kS / kSeg;            // K-split for mix1
constexpr float kScale = 22.627416997969522f;           // sqrt(512)

__device__ inline bf16 f2bf(float f) { return (bf16)f; }  // native v_cvt, RNE
__device__ inline bf16 u16bf(unsigned short s) {
  union { unsigned short s; bf16 b; } u; u.s = s; return u.b;
}
__device__ inline float gelu_erf(float x) {
  return 0.5f * x * (1.0f + erff(x * 0.7071067811865476f));
}

// LDS byte offset from a generic pointer: flat aperture maps LDS to addr[31:0].
__device__ inline unsigned lds_off(const void* p) {
  return (unsigned)(size_t)p;
}
// Async global->LDS 16B copy (per-lane), tracked by ASYNCcnt.
__device__ inline void async_ld_b128(unsigned lds, unsigned goff,
                                     const void* sbase) {
  asm volatile("global_load_async_to_lds_b128 %0, %1, %2"
               :: "v"(lds), "v"(goff), "s"(sbase) : "memory");
}
__device__ inline void wait_async0() {
  asm volatile("s_wait_asynccnt 0x0" ::: "memory");
}

// A fragment (16x32 bf16, MxK): lane<16 -> M=lane, K in {0..15};
// lane>=16 -> same M, K+16. K-pairs contiguous -> 8 dword LDS loads.
template <int STR>
__device__ inline v16bf load_a_frag(const bf16* tile, int mrow, int k0, int hl) {
  union { v16bf v; unsigned u[8]; } f;
  const bf16* r = tile + mrow * STR + k0 + 16 * hl;
#pragma unroll
  for (int i = 0; i < 8; ++i) {
    int kb = 2 * i;
    f.u[i] = *(const unsigned*)(r + kb);
  }
  return f.v;
}
// B fragment (32x16 bf16, KxN): lane = column, K = 16*half + e.
// Tile stored per-column-K-contiguous -> single v16bf (2x b128) LDS load.
template <int STR>
__device__ inline v16bf load_b_frag(const bf16* tile, int ncol, int k0, int hl) {
  return *(const v16bf*)(tile + ncol * STR + k0 + 16 * hl);
}
__device__ inline v8f wmma_bf16(v16bf a, v16bf b, v8f c) {
  return __builtin_amdgcn_wmma_f32_16x16x32_bf16(false, a, false, b, (short)0,
                                                 c, false, false);
}

// ---------------------------------------------------------------------------
// Kernel 1: W = fc2^T( gelu( fc1^T( src*scale + pe ) ) ) per head slice.
// grid (S/64, B*H), block 128 (4 waves).
// ---------------------------------------------------------------------------
__global__ __launch_bounds__(128) void k_build_w(
    const float* __restrict__ src, const float* __restrict__ pe,
    const float* __restrict__ fc1, const float* __restrict__ fc2,
    bf16* __restrict__ Wout) {
  const int bh = blockIdx.y, b = bh >> 3, hd = bh & 7;
  const int s0 = blockIdx.x * 64;
  const int t = threadIdx.x, wv = t >> 5, lane = t & 31;
  const int hl = lane >> 4, ln = lane & 15;

  __shared__ alignas(32) bf16 x_lds[64 * 72];
  __shared__ alignas(32) bf16 fc1_lds[64 * 64];   // [n][k]
  __shared__ alignas(32) bf16 fc2_lds[256 * 64];  // [n][k]
  __shared__ alignas(32) bf16 h_lds[64 * 72];

  for (int i = t; i < 64 * 64; i += 128) {
    int r = i >> 6, c = i & 63;
    float v = src[(size_t)(b * kS + s0 + r) * kD + hd * kDH + c] * kScale +
              pe[(size_t)(s0 + r) * kD + hd * kDH + c];
    x_lds[r * 72 + c] = f2bf(v);
  }
  const float* f1 = fc1 + (size_t)hd * kDH * kDH;
  for (int i = t; i < kDH * kDH; i += 128) fc1_lds[i] = f2bf(f1[i]);
  const float* f2p = fc2 + (size_t)hd * kHYPH * kDH;
  for (int i = t; i < kHYPH * kDH; i += 128) fc2_lds[i] = f2bf(f2p[i]);
  __syncthreads();

  const int mw = wv * 16;

  // h = gelu( X(16x64) @ fc1^T )
  v8f acc[4] = {};
#pragma unroll
  for (int k0 = 0; k0 < 64; k0 += 32) {
    v16bf a = load_a_frag<72>(x_lds, mw + ln, k0, hl);
#pragma unroll
    for (int j = 0; j < 4; ++j) {
      v16bf bb = load_b_frag<64>(fc1_lds, j * 16 + ln, k0, hl);
      acc[j] = wmma_bf16(a, bb, acc[j]);
    }
  }
#pragma unroll
  for (int j = 0; j < 4; ++j)
#pragma unroll
    for (int r = 0; r < 8; ++r)
      h_lds[(mw + r + 8 * hl) * 72 + j * 16 + ln] = f2bf(gelu_erf(acc[j][r]));
  __syncthreads();

  // W = h(16x64) @ fc2^T(64x256), 4 chunks of 64 cols
#pragma unroll
  for (int ch = 0; ch < 4; ++ch) {
    v8f a2[4] = {};
#pragma unroll
    for (int k0 = 0; k0 < 64; k0 += 32) {
      v16bf a = load_a_frag<72>(h_lds, mw + ln, k0, hl);
#pragma unroll
      for (int j = 0; j < 4; ++j) {
        v16bf bb = load_b_frag<64>(fc2_lds, ch * 64 + j * 16 + ln, k0, hl);
        a2[j] = wmma_bf16(a, bb, a2[j]);
      }
    }
#pragma unroll
    for (int j = 0; j < 4; ++j)
#pragma unroll
      for (int r = 0; r < 8; ++r) {
        int row = s0 + mw + r + 8 * hl;
        int col = ch * 64 + j * 16 + ln;
        Wout[((size_t)bh * kS + row) * kHYPH + col] = f2bf(a2[j][r]);
      }
  }
}

// ---------------------------------------------------------------------------
// Kernel 2: partial O = V(64xKseg) @ W1(Ksegx256), K-split over blockIdx.y.
// grid (B*H, kSeg), block 256 (8 waves). Writes fp32 partials (no gelu yet).
// ---------------------------------------------------------------------------
__global__ __launch_bounds__(256) void k_mix1(
    const float* __restrict__ value, const bf16* __restrict__ W1,
    float* __restrict__ Opart) {
  const int bh = blockIdx.x, b = bh >> 3, hd = bh & 7;
  const int seg = blockIdx.y;
  const int t = threadIdx.x, wv = t >> 5, lane = t & 31;
  const int hl = lane >> 4, ln = lane & 15;
  const int mt = (wv & 3) * 16, nb = (wv >> 2) * 128;

  __shared__ alignas(32) bf16 V_lds[64 * 36];    // [d][s_local]
  __shared__ alignas(32) bf16 B_ldsT[256 * 48];  // [n][k_local]

  v8f acc[8] = {};
  for (int sK = seg * kSegLen; sK < (seg + 1) * kSegLen; sK += 32) {
    __syncthreads();
    {  // stage V tile (64 d x 32 s), transposed from (s, d) global order
      int sl = t >> 3, dg = (t & 7) * 8;
      const float* vp = value + (size_t)(b * kS + sK + sl) * kD + hd * kDH + dg;
#pragma unroll
      for (int i = 0; i < 8; ++i) V_lds[(dg + i) * 36 + sl] = f2bf(vp[i]);
    }
    {  // stage W1 tile (32 k x 256 n) transposed -> [n][k]
      int row = t >> 3, cb = (t & 7) * 32;
      const unsigned* wp =
          (const unsigned*)(W1 + ((size_t)bh * kS + sK + row) * kHYPH + cb);
#pragma unroll
      for (int i = 0; i < 16; ++i) {
        unsigned v = wp[i];
        B_ldsT[(cb + 2 * i) * 48 + row] = u16bf((unsigned short)(v & 0xffffu));
        B_ldsT[(cb + 2 * i + 1) * 48 + row] = u16bf((unsigned short)(v >> 16));
      }
    }
    __syncthreads();
    v16bf a = load_a_frag<36>(V_lds, mt + ln, 0, hl);
#pragma unroll
    for (int j = 0; j < 8; ++j) {
      v16bf bb = load_b_frag<48>(B_ldsT, nb + j * 16 + ln, 0, hl);
      acc[j] = wmma_bf16(a, bb, acc[j]);
    }
  }
  float* op = Opart + ((size_t)seg * (kB * kH) + bh) * (kDH * kHYPH);
#pragma unroll
  for (int j = 0; j < 8; ++j)
#pragma unroll
    for (int r = 0; r < 8; ++r)
      op[(mt + r + 8 * hl) * kHYPH + nb + j * 16 + ln] = acc[j][r];
}

// Finalize: O = bf16(gelu(sum of kSeg partials)). 64*64*256 elements.
__global__ __launch_bounds__(256) void k_ofin(const float* __restrict__ Opart,
                                              bf16* __restrict__ O) {
  const int i = blockIdx.x * 256 + threadIdx.x;
  constexpr int N = kB * kH * kDH * kHYPH;
  float s = 0.f;
#pragma unroll
  for (int g = 0; g < kSeg; ++g) s += Opart[(size_t)g * N + i];
  O[i] = f2bf(gelu_erf(s));
}

// ---------------------------------------------------------------------------
// Kernel 3: out_pre = O(64x256) @ W2^T(256x2048) -> (B,S,D) fp32.
// grid (S/128, B*H), block 128. Tiles staged with async global->LDS b128.
// ---------------------------------------------------------------------------
__global__ __launch_bounds__(128) void k_mix2(
    const bf16* __restrict__ O, const bf16* __restrict__ W2,
    float* __restrict__ out) {
  const int bh = blockIdx.y, b = bh >> 3, hd = bh & 7;
  const int s0 = blockIdx.x * 128;
  const int t = threadIdx.x, wv = t >> 5, lane = t & 31;
  const int hl = lane >> 4, ln = lane & 15;
  const int mt = wv * 16;

  __shared__ alignas(32) bf16 O_lds[64 * 264];  // [d][k], row stride 528 B
  __shared__ alignas(32) bf16 B_lds[128 * 48];  // [n][k_local], row 96 B

  {  // async-stage O (64 x 256 bf16 = 32 KB): 2048 x 16B chunks
    const void* obase = (const void*)(O + (size_t)bh * (kDH * kHYPH));
    const unsigned l0 = lds_off(&O_lds[0]);
    for (int f = t; f < 2048; f += 128) {
      int r = f >> 5, c = f & 31;
      async_ld_b128(l0 + (unsigned)(r * 528 + c * 16),
                    (unsigned)(r * 512 + c * 16), obase);
    }
  }
  const void* wbase = (const void*)(W2 + ((size_t)bh * kS + s0) * kHYPH);
  const unsigned lB = lds_off(&B_lds[0]) + (unsigned)(t * 96);

  v8f acc[8] = {};
  for (int k0 = 0; k0 < kHYPH; k0 += 32) {
    __syncthreads();  // prior wmma done reading B_lds before overwrite
    {  // async-stage W2^T tile: B[n][k] = W2[s0+n][k0+k], 4 x 16B per lane
      unsigned goff = (unsigned)(t * 512 + k0 * 2);
#pragma unroll
      for (int i = 0; i < 4; ++i)
        async_ld_b128(lB + 16u * i, goff + 16u * i, wbase);
    }
    wait_async0();
    __syncthreads();
    v16bf a = load_a_frag<264>(O_lds, mt + ln, k0, hl);
#pragma unroll
    for (int j = 0; j < 8; ++j) {
      v16bf bb = load_b_frag<48>(B_lds, j * 16 + ln, 0, hl);
      acc[j] = wmma_bf16(a, bb, acc[j]);
    }
  }
#pragma unroll
  for (int j = 0; j < 8; ++j)
#pragma unroll
    for (int r = 0; r < 8; ++r) {
      int s = s0 + j * 16 + ln;
      int ch = hd * kDH + mt + r + 8 * hl;
      out[((size_t)(b * kS + s)) * kD + ch] = acc[j][r];
    }
}

// ---------------------------------------------------------------------------
// Kernel 4: in-place LayerNorm over D=512. One block (256 thr) per row.
// ---------------------------------------------------------------------------
__global__ __launch_bounds__(256) void k_ln(float* __restrict__ out,
                                            const float* __restrict__ gamma,
                                            const float* __restrict__ beta) {
  const int t = threadIdx.x;
  float* p = out + (size_t)blockIdx.x * kD;
  float x0 = p[t], x1 = p[t + 256];
  __shared__ float s1[256], s2[256];
  s1[t] = x0 + x1;
  s2[t] = x0 * x0 + x1 * x1;
  __syncthreads();
  for (int off = 128; off > 0; off >>= 1) {
    if (t < off) { s1[t] += s1[t + off]; s2[t] += s2[t + off]; }
    __syncthreads();
  }
  float mu = s1[0] * (1.0f / kD);
  float var = s2[0] * (1.0f / kD) - mu * mu;
  float rs = rsqrtf(var + 1e-5f);
  p[t]       = (x0 - mu) * rs * gamma[t]       + beta[t];
  p[t + 256] = (x1 - mu) * rs * gamma[t + 256] + beta[t + 256];
}

// ---------------------------------------------------------------------------
extern "C" void kernel_launch(void* const* d_in, const int* in_sizes, int n_in,
                              void* d_out, int out_size, void* d_ws,
                              size_t ws_size, hipStream_t stream) {
  (void)in_sizes; (void)n_in; (void)out_size; (void)ws_size;
  const float* query  = (const float*)d_in[0];
  const float* key_t  = (const float*)d_in[1];
  const float* value  = (const float*)d_in[2];
  const float* w1_fc1 = (const float*)d_in[3];
  const float* w1_fc2 = (const float*)d_in[4];
  const float* w2_fc1 = (const float*)d_in[5];
  const float* w2_fc2 = (const float*)d_in[6];
  const float* ln_g   = (const float*)d_in[7];
  const float* ln_b   = (const float*)d_in[8];
  const float* pe     = (const float*)d_in[9];
  float* out = (float*)d_out;

  // ws: W1 64MiB | W2 64MiB | O_bf16 2MiB | Opart fp32 4x4MiB  (~146 MB)
  char* ws = (char*)d_ws;
  const size_t wbytes = (size_t)kB * kH * kS * kHYPH * sizeof(bf16);
  const size_t obytes = (size_t)kB * kH * kDH * kHYPH * sizeof(bf16);
  bf16*  W1    = (bf16*)ws;
  bf16*  W2    = (bf16*)(ws + wbytes);
  bf16*  O     = (bf16*)(ws + 2 * wbytes);
  float* Opart = (float*)(ws + 2 * wbytes + obytes);

  dim3 gW(kS / 64, kB * kH);
  k_build_w<<<gW, 128, 0, stream>>>(key_t, pe, w1_fc1, w1_fc2, W1);
  k_build_w<<<gW, 128, 0, stream>>>(query, pe, w2_fc1, w2_fc2, W2);
  k_mix1<<<dim3(kB * kH, kSeg), 256, 0, stream>>>(value, W1, Opart);
  k_ofin<<<dim3(kB * kH * kDH * kHYPH / 256), 256, 0, stream>>>(Opart, O);
  k_mix2<<<dim3(kS / 128, kB * kH), 128, 0, stream>>>(O, W2, out);
  k_ln<<<dim3(kB * kS), 256, 0, stream>>>(out, ln_g, ln_b);
}
